// QuantizeLatent_14980845928565
// MI455X (gfx1250) — matrix operand order repaired and checked by reading, not compiled, not run on hardware
//
#include <hip/hip_runtime.h>

typedef __attribute__((ext_vector_type(16))) _Float16 v16h;
typedef __attribute__((ext_vector_type(8)))  _Float16 v8h;
typedef __attribute__((ext_vector_type(8)))  float    v8f;

#define N_ROWS   (32 * 4096)   // 131072
#define CDIM     64
#define KCODES   1024
#define BETA_F   0.1f

#define WAVES_PER_BLOCK 4
#define ROWTILES        2                       // 16-row WMMA tiles per wave
#define ROWS_PER_WAVE   (16 * ROWTILES)         // 32
#define MAIN_BLOCKS     (N_ROWS / (ROWS_PER_WAVE * WAVES_PER_BLOCK))  // 1024

// ---------------------------------------------------------------------------
// Prep: emb f32 -> f16 copy + per-code squared norms. One wave per code.
// ---------------------------------------------------------------------------
__global__ __launch_bounds__(128)
void vq_prep_kernel(const float* __restrict__ emb,
                    _Float16* __restrict__ embh,
                    float* __restrict__ enorm) {
    int tid  = blockIdx.x * blockDim.x + threadIdx.x;
    int code = tid >> 5;
    int lane = tid & 31;
    if (code >= KCODES) return;
    const float* row = emb + (size_t)code * CDIM;
    float x = row[lane * 2 + 0];
    float y = row[lane * 2 + 1];
    embh[(size_t)code * CDIM + lane * 2 + 0] = (_Float16)x;
    embh[(size_t)code * CDIM + lane * 2 + 1] = (_Float16)y;
    float s = x * x + y * y;
#pragma unroll
    for (int m = 16; m >= 1; m >>= 1) s += __shfl_xor(s, m, 32);
    if (lane == 0) enorm[code] = s;
}

// ---------------------------------------------------------------------------
// Main: per wave, 32 rows vs all 1024 codes via v_wmma_f32_16x16x32_f16.
// dist = 2*(z.e) + ||e||^2  (the ||z||^2 term is row-constant -> argmin-safe)
// ---------------------------------------------------------------------------
__global__ __launch_bounds__(128)
void vq_main_kernel(const float* __restrict__ z,
                    const float* __restrict__ emb,
                    const _Float16* __restrict__ embh,
                    const float* __restrict__ enorm,
                    float* __restrict__ out,
                    float* __restrict__ partials) {
    const int lane    = threadIdx.x & 31;
    const int wib     = threadIdx.x >> 5;
    const int waveId  = blockIdx.x * WAVES_PER_BLOCK + wib;
    const int rowBase = waveId * ROWS_PER_WAVE;
    const int half    = lane >> 4;   // lane half selects K sub-slices
    const int m       = lane & 15;   // A: row within tile, B: column (code)

    // ---- Load A tiles (z rows) once, f32 -> f16, ISA 16-bit A layout:
    // lanes 0-15 : K = {cb+0..7,  cb+16..23}, lanes 16-31: K = {cb+8..15, cb+24..31}
    v16h a[ROWTILES][2];
#pragma unroll
    for (int t = 0; t < ROWTILES; ++t) {
        const float* zr = z + (size_t)(rowBase + t * 16 + m) * CDIM;
#pragma unroll
        for (int c = 0; c < 2; ++c) {
            const int kb = c * 32 + half * 8;
            float4 p0 = *(const float4*)(zr + kb);
            float4 p1 = *(const float4*)(zr + kb + 4);
            float4 p2 = *(const float4*)(zr + kb + 16);
            float4 p3 = *(const float4*)(zr + kb + 20);
            v16h av;
            av[0]  = (_Float16)p0.x; av[1]  = (_Float16)p0.y;
            av[2]  = (_Float16)p0.z; av[3]  = (_Float16)p0.w;
            av[4]  = (_Float16)p1.x; av[5]  = (_Float16)p1.y;
            av[6]  = (_Float16)p1.z; av[7]  = (_Float16)p1.w;
            av[8]  = (_Float16)p2.x; av[9]  = (_Float16)p2.y;
            av[10] = (_Float16)p2.z; av[11] = (_Float16)p2.w;
            av[12] = (_Float16)p3.x; av[13] = (_Float16)p3.y;
            av[14] = (_Float16)p3.z; av[15] = (_Float16)p3.w;
            a[t][c] = av;
        }
    }

    float bestVal[ROWTILES][8];
    int   bestIdx[ROWTILES][8];
#pragma unroll
    for (int t = 0; t < ROWTILES; ++t)
#pragma unroll
        for (int i = 0; i < 8; ++i) { bestVal[t][i] = 3.4e38f; bestIdx[t][i] = 0; }

    // ---- Sweep the codebook in 16-code tiles.
    for (int ct = 0; ct < KCODES / 16; ++ct) {
        const int codeBase = ct << 4;
        // B layout (dense 16-bit 32x16): lane holds column N=m; lanes 0-15: K {0-7,16-23},
        // lanes 16-31: K {8-15,24-31} (same K order as A).
        const _Float16* er = embh + (size_t)(codeBase + m) * CDIM + half * 8;
        v8h b0lo = *(const v8h*)(er);
        v8h b0hi = *(const v8h*)(er + 16);
        v8h b1lo = *(const v8h*)(er + 32);
        v8h b1hi = *(const v8h*)(er + 48);
        v16h b0, b1;
#pragma unroll
        for (int q = 0; q < 8; ++q) {
            b0[q] = b0lo[q]; b0[q + 8] = b0hi[q];
            b1[q] = b1lo[q]; b1[q + 8] = b1hi[q];
        }
        const float en   = enorm[codeBase + m];
        const int   cidx = codeBase + m;

#pragma unroll
        for (int t = 0; t < ROWTILES; ++t) {
            v8f acc = {};
            acc = __builtin_amdgcn_wmma_f32_16x16x32_f16(
                false, a[t][0], false, b0, (short)0, acc, false, false);
            acc = __builtin_amdgcn_wmma_f32_16x16x32_f16(
                false, a[t][1], false, b1, (short)0, acc, false, false);
            // C/D layout: slot i in this lane = S[M = i + 8*half][N = m]
#pragma unroll
            for (int i = 0; i < 8; ++i) {
                float d = 2.0f * acc[i] + en;
                if (d < bestVal[t][i]) { bestVal[t][i] = d; bestIdx[t][i] = cidx; }
            }
        }
    }

    // ---- Reduce argmin across the 16 lanes of each half (the N dimension),
    // first-index tie-break to match jnp.argmin.
#pragma unroll
    for (int t = 0; t < ROWTILES; ++t)
#pragma unroll
        for (int i = 0; i < 8; ++i)
#pragma unroll
            for (int msk = 1; msk < 16; msk <<= 1) {
                float ov = __shfl_xor(bestVal[t][i], msk, 32);
                int   oi = __shfl_xor(bestIdx[t][i], msk, 32);
                if (ov < bestVal[t][i] ||
                    (ov == bestVal[t][i] && oi < bestIdx[t][i])) {
                    bestVal[t][i] = ov; bestIdx[t][i] = oi;
                }
            }

    // ---- Gather zq = emb[idx] (f32), write out, accumulate (zq - z)^2.
    float lsum = 0.0f;
#pragma unroll
    for (int t = 0; t < ROWTILES; ++t) {
#pragma unroll
        for (int r = 0; r < 16; ++r) {
            const int i   = r & 7;
            const int src = (r < 8) ? 0 : 16;     // half that owns row r
            const int q   = __shfl(bestIdx[t][i], src, 32);
            const size_t row = (size_t)(rowBase + t * 16 + r);
            float2 e2 = *(const float2*)(emb + (size_t)q * CDIM + lane * 2);
            float2 z2 = *(const float2*)(z + row * CDIM + lane * 2);
            *(float2*)(out + row * CDIM + lane * 2) = e2;
            float dx = e2.x - z2.x, dy = e2.y - z2.y;
            lsum += dx * dx + dy * dy;
        }
    }

#pragma unroll
    for (int msk = 16; msk >= 1; msk >>= 1) lsum += __shfl_xor(lsum, msk, 32);

    __shared__ float wsum[WAVES_PER_BLOCK];
    if (lane == 0) wsum[wib] = lsum;
    __syncthreads();
    if (threadIdx.x == 0) {
        float s = 0.0f;
        for (int w = 0; w < WAVES_PER_BLOCK; ++w) s += wsum[w];
        partials[blockIdx.x] = s;
    }
}

// ---------------------------------------------------------------------------
// Deterministic final reduction -> scalar loss.
// ---------------------------------------------------------------------------
__global__ __launch_bounds__(256)
void vq_finish_kernel(const float* __restrict__ partials,
                      float* __restrict__ loss_out) {
    __shared__ float sm[256];
    float s = 0.0f;
    for (int i = threadIdx.x; i < MAIN_BLOCKS; i += 256) s += partials[i];
    sm[threadIdx.x] = s;
    __syncthreads();
    for (int off = 128; off >= 1; off >>= 1) {
        if ((int)threadIdx.x < off) sm[threadIdx.x] += sm[threadIdx.x + off];
        __syncthreads();
    }
    if (threadIdx.x == 0)
        loss_out[0] = sm[0] * (1.0f + BETA_F) / (float)((size_t)N_ROWS * CDIM);
}

// ---------------------------------------------------------------------------
extern "C" void kernel_launch(void* const* d_in, const int* in_sizes, int n_in,
                              void* d_out, int out_size, void* d_ws, size_t ws_size,
                              hipStream_t stream) {
    const float* z   = (const float*)d_in[0];   // [32,4096,64] f32
    const float* emb = (const float*)d_in[1];   // [1024,64] f32
    float* out       = (float*)d_out;           // [N_ROWS*CDIM] + 1 (loss)
    float* loss_out  = out + (size_t)N_ROWS * CDIM;

    char* ws = (char*)d_ws;
    _Float16* embh  = (_Float16*)ws;                                    // 128 KB
    float*    enorm = (float*)(ws + (size_t)KCODES * CDIM * 2);         // 4 KB
    float*    parts = enorm + KCODES;                                   // 4 KB

    vq_prep_kernel<<<KCODES / WAVES_PER_BLOCK, 128, 0, stream>>>(emb, embh, enorm);
    vq_main_kernel<<<MAIN_BLOCKS, 128, 0, stream>>>(z, emb, embh, enorm, out, parts);
    vq_finish_kernel<<<1, 256, 0, stream>>>(parts, loss_out);
}
